// CompGCNCov_49452253446794
// MI455X (gfx1250) — matrix operand order repaired and checked by reading, not compiled, not run on hardware
//
#include <hip/hip_runtime.h>
#include <hip/hip_bf16.h>

typedef __attribute__((ext_vector_type(16))) __bf16 v16bf;
typedef __attribute__((ext_vector_type(8)))  __bf16 v8bf;
typedef __attribute__((ext_vector_type(4)))  __bf16 v4bf;
typedef __attribute__((ext_vector_type(8)))  float  v8f;
typedef __attribute__((ext_vector_type(4)))  int    v4i;

#define D        256
#define NNODES   80000
#define NEDGES   800000
#define HALFE    400000
#define NREL     400
#define NBINS    800          // dir*400 + type
#define MAT_EL   65536        // 256*256
#define AS_STR   264          // padded LDS row stride (elems) to dodge bank conflicts
#define MAX_TILES (NEDGES/16 + NBINS)   // 50800 upper bound on Σ ceil(cnt/16)

#define HAS_ASYNC_LDS (__has_builtin(__builtin_amdgcn_global_load_async_to_lds_b128))

// ---------------- async global->LDS 16B copy (CDNA5 DMA path) ----------------
__device__ __forceinline__ void async_copy16(const void* g, void* l) {
#if HAS_ASYNC_LDS
    __builtin_amdgcn_global_load_async_to_lds_b128(
        (__attribute__((address_space(1))) v4i*)g,
        (__attribute__((address_space(3))) v4i*)l, 0, 0);
#else
    *(v8bf*)l = *(const v8bf*)g;
#endif
}
__device__ __forceinline__ void async_wait_all() {
#if HAS_ASYNC_LDS
#if __has_builtin(__builtin_amdgcn_s_wait_asynccnt)
    __builtin_amdgcn_s_wait_asynccnt(0);
#else
    asm volatile("s_wait_asynccnt 0x0" ::: "memory");
#endif
#endif
}

// ---------------- WMMA fragment helpers (wave32, 16x16x32 bf16) ----------------
// A 16x32 (MxK) bf16: lane<16 -> M=lane,    K = {0..7, 16..23}
//                     lane>=16 -> M=lane-16, K = {8..15, 24..31}
__device__ __forceinline__ v16bf load_a_lds(const __bf16* base, int stride,
                                            int k0, int lane) {
    const int m  = lane & 15;
    const int kb = (lane < 16) ? 0 : 8;
    const __bf16* row = base + m * stride + k0 + kb;
    v16bf a;
#pragma unroll
    for (int i = 0; i < 8; ++i) a[i] = row[i];
#pragma unroll
    for (int i = 0; i < 8; ++i) a[8 + i] = row[16 + i];
    return a;
}

// B 32x16 (KxN) bf16 from an N-major (transposed) matrix BT[n][k]:
// lane<16 -> N=lane, K={0..7,16..23} ; lane>=16 -> N=lane-16, K={8..15,24..31}.
__device__ __forceinline__ v16bf load_b_nt(const __bf16* BT, int ldk,
                                           int n0, int k0, int lane) {
    const int n  = n0 + (lane & 15);
    const int kb = (lane < 16) ? 0 : 8;
    const __bf16* p = BT + (size_t)n * ldk + k0 + kb;
    const v8bf lo = *(const v8bf*)p;
    const v8bf hi = *(const v8bf*)(p + 16);
    return __builtin_shufflevector(lo, hi, 0,1,2,3,4,5,6,7,8,9,10,11,12,13,14,15);
}

__device__ __forceinline__ v8f wmma_bf16(v16bf a, v16bf b, v8f c) {
    return __builtin_amdgcn_wmma_f32_16x16x32_bf16(false, a, false, b,
                                                   (short)0, c, false, false);
}

// ---------------- utility kernels ----------------
__global__ void k_zero32(unsigned* p, long long n) {
    long long i = (long long)blockIdx.x * blockDim.x + threadIdx.x;
    if (i < n) p[i] = 0u;
}

// x (f32) -> Xb (bf16), vector-4 per thread
__global__ void k_x_to_bf16(const float* x, __bf16* xb) {
    const size_t i = ((size_t)blockIdx.x * 256 + threadIdx.x) * 4;
    const float4 v = *(const float4*)(x + i);
    v4bf o;
    o[0] = (__bf16)v.x; o[1] = (__bf16)v.y; o[2] = (__bf16)v.z; o[3] = (__bf16)v.w;
    *(v4bf*)(xb + i) = o;
}

// WT[mat][n][k] = W[k][n] as bf16; mats: 0=in_w 1=out_w 2=loop_w 3=w_rel
__global__ void k_transpose_w(const float* in_w, const float* out_w,
                              const float* loop_w, const float* w_rel,
                              __bf16* WT) {
    const int mat = blockIdx.x >> 8;
    const int n   = blockIdx.x & 255;
    const int k   = threadIdx.x;
    const float* W = (mat == 0) ? in_w : (mat == 1) ? out_w
                    : (mat == 2) ? loop_w : w_rel;
    WT[(size_t)mat * MAT_EL + (size_t)n * D + k] = (__bf16)W[(size_t)k * D + n];
}

// PT[bin][n][j] = Σ_k rel[(j+k)%256] * W[k][n]   (bin 0..799 edges, 800 = loop)
__global__ void k_build_composite(const float* rel_repr, const float* loop_rel,
                                  const __bf16* WT, __bf16* PT) {
    __shared__ __align__(16) __bf16 rel2[512];
    const int bin = blockIdx.x;
    const float*  rv;
    const __bf16* wt;
    if (bin < NREL)          { rv = rel_repr + (size_t)bin * D;          wt = WT; }
    else if (bin < 2 * NREL) { rv = rel_repr + (size_t)(bin - NREL) * D; wt = WT + MAT_EL; }
    else                     { rv = loop_rel;                            wt = WT + 2 * MAT_EL; }

    const int tid = threadIdx.x;                 // 256 threads = 8 waves
    const float v = rv[tid];
    rel2[tid] = (__bf16)v; rel2[tid + 256] = (__bf16)v;
    __syncthreads();

    const int wave = tid >> 5, lane = tid & 31;
    const int ln   = lane & 15;
    const int kb_a = (lane < 16) ? 0 : 8;
    __bf16* PTb = PT + (size_t)bin * MAT_EL;

    for (int ti = 0; ti < 32; ++ti) {            // each wave: 32 of 256 tiles
        const int t  = wave * 32 + ti;
        const int mt = t >> 4, nt = t & 15;
        v8f acc = {};
#pragma unroll
        for (int k0 = 0; k0 < D; k0 += 32) {
            // A[j][k] = rel[(j+k)%256] -> duplicated rel2, no mask needed
            const __bf16* row = rel2 + (mt * 16 + ln) + k0 + kb_a;
            v16bf a;
#pragma unroll
            for (int i = 0; i < 8; ++i) a[i] = row[i];
#pragma unroll
            for (int i = 0; i < 8; ++i) a[8 + i] = row[16 + i];
            const v16bf b = load_b_nt(wt, D, nt * 16, k0, lane);
            acc = wmma_bf16(a, b, acc);
        }
        // store transposed: PT[n][j], 8 consecutive j per lane
        const int n  = nt * 16 + ln;
        const int j0 = mt * 16 + ((lane < 16) ? 0 : 8);
        v8bf o;
#pragma unroll
        for (int r = 0; r < 8; ++r) o[r] = (__bf16)acc[r];
        *(v8bf*)(PTb + (size_t)n * D + j0) = o;
    }
}

// ---------------- counting sort by (dir, type) ----------------
__global__ void k_hist(const int* etype, int* hist) {
    const int e = blockIdx.x * 256 + threadIdx.x;
    if (e >= NEDGES) return;
    atomicAdd(&hist[((e < HALFE) ? 0 : NREL) + etype[e]], 1);
}

__global__ void k_scan(const int* hist, int* offsets, int* cursor, int* tileoff) {
    __shared__ int s[1024];
    const int t = threadIdx.x;
    const int c = (t < NBINS) ? hist[t] : 0;
    s[t] = c; __syncthreads();
    for (int d = 1; d < 1024; d <<= 1) {
        const int v = (t >= d) ? s[t - d] : 0; __syncthreads();
        s[t] += v; __syncthreads();
    }
    if (t < NBINS) { offsets[t] = s[t] - c; cursor[t] = s[t] - c; }
    if (t == NBINS - 1) offsets[NBINS] = s[t];
    __syncthreads();
    const int tc = (t < NBINS) ? ((c + 15) >> 4) : 0;
    s[t] = tc; __syncthreads();
    for (int d = 1; d < 1024; d <<= 1) {
        const int v = (t >= d) ? s[t - d] : 0; __syncthreads();
        s[t] += v; __syncthreads();
    }
    if (t < NBINS) tileoff[t] = s[t] - tc;
    if (t == NBINS - 1) tileoff[NBINS] = s[t];
}

__global__ void k_scatter_idx(const int* etype, int* cursor, int* sorted) {
    const int e = blockIdx.x * 256 + threadIdx.x;
    if (e >= NEDGES) return;
    const int bin = ((e < HALFE) ? 0 : NREL) + etype[e];
    sorted[atomicAdd(&cursor[bin], 1)] = e;
}

// ---------------- main edge GEMM: async gather -> WMMA -> atomic scatter ----------------
__global__ void k_edge_gemm(const __bf16* Xb, const int* esrc, const int* edst,
                            const float* enorm, const int* sorted,
                            const int* offsets, const int* tileoff,
                            const __bf16* PT, float* out) {
    __shared__ __align__(16) __bf16 As[16 * AS_STR];
    __shared__ int   s_src[16];
    __shared__ int   s_dst[16];
    __shared__ float s_nrm[16];

    const int tile  = blockIdx.x;
    if (tile >= tileoff[NBINS]) return;          // uniform exit

    int lo = 0, hi = NBINS;                      // find bin: tileoff[b] <= tile
    while (hi - lo > 1) { const int mid = (lo + hi) >> 1;
                          if (tileoff[mid] <= tile) lo = mid; else hi = mid; }
    const int bin   = lo;
    const int ebase = offsets[bin] + (tile - tileoff[bin]) * 16;
    int ecnt = offsets[bin + 1] - ebase; if (ecnt > 16) ecnt = 16;

    const int tid = threadIdx.x;                 // 128 threads = 4 waves
    if (tid < 16) {
        int src = 0, dst = 0; float nm = 0.f;
        if (tid < ecnt) {
            const int e = sorted[ebase + tid];
            src = esrc[e]; dst = edst[e]; nm = enorm[e] * (1.f / 3.f);
        }
        s_src[tid] = src; s_dst[tid] = dst; s_nrm[tid] = nm;
    }
    __syncthreads();
    // async DMA gather: 16 rows x 512B (bf16), 16B per lane-op, 512 chunks
    {
        char* Asb = (char*)As;
        for (int i = tid; i < 16 * 32; i += 128) {
            const int m = i >> 5, c = i & 31;
            async_copy16(Xb + (size_t)s_src[m] * D + c * 8,
                         Asb + (size_t)m * (AS_STR * 2) + c * 16);
        }
        async_wait_all();
    }
    __syncthreads();

    const int wave = tid >> 5, lane = tid & 31;
    const int n0   = wave * 64;                  // 4 n-tiles per wave
    const __bf16* PTb = PT + (size_t)bin * MAT_EL;
    v8f acc[4] = {};
#pragma unroll
    for (int k0 = 0; k0 < D; k0 += 32) {
        const v16bf a = load_a_lds(As, AS_STR, k0, lane);
        __builtin_prefetch(PTb + (size_t)(n0 + (lane & 15)) * D + k0 + 64, 0, 1);
#pragma unroll
        for (int nt = 0; nt < 4; ++nt) {
            const v16bf b = load_b_nt(PTb, D, n0 + nt * 16, k0, lane);
            acc[nt] = wmma_bf16(a, b, acc[nt]);
        }
    }
    const int m0 = (lane < 16) ? 0 : 8;
#pragma unroll
    for (int nt = 0; nt < 4; ++nt) {
        const int n = n0 + nt * 16 + (lane & 15);
#pragma unroll
        for (int r = 0; r < 8; ++r) {
            const int m = m0 + r;
            atomicAdd(&out[(size_t)s_dst[m] * D + n], acc[nt][r] * s_nrm[m]);
        }
    }
}

// ---------------- self-loop: out += (x @ P_loop)/3 + bias ----------------
__global__ void k_loop_gemm(const __bf16* Xb, const __bf16* PT_loop,
                            const float* bias, float* out) {
    __shared__ __align__(16) __bf16 As[16 * AS_STR];
    const int mbase = blockIdx.x * 16;
    const int tid   = threadIdx.x;               // 128
    {
        char* Asb = (char*)As;
        for (int i = tid; i < 16 * 32; i += 128) {
            const int m = i >> 5, c = i & 31;
            async_copy16(Xb + (size_t)(mbase + m) * D + c * 8,
                         Asb + (size_t)m * (AS_STR * 2) + c * 16);
        }
        async_wait_all();
    }
    __syncthreads();
    const int wave = tid >> 5, lane = tid & 31;
    const int n0   = wave * 64;
    v8f acc[4] = {};
#pragma unroll
    for (int k0 = 0; k0 < D; k0 += 32) {
        const v16bf a = load_a_lds(As, AS_STR, k0, lane);
#pragma unroll
        for (int nt = 0; nt < 4; ++nt) {
            const v16bf b = load_b_nt(PT_loop, D, n0 + nt * 16, k0, lane);
            acc[nt] = wmma_bf16(a, b, acc[nt]);
        }
    }
    const int m0 = (lane < 16) ? 0 : 8;
#pragma unroll
    for (int nt = 0; nt < 4; ++nt) {
        const int n = n0 + nt * 16 + (lane & 15);
#pragma unroll
        for (int r = 0; r < 8; ++r) {
            const size_t idx = (size_t)(mbase + m0 + r) * D + n;
            out[idx] += acc[nt][r] * (1.f / 3.f) + bias[n];
        }
    }
}

// ---------------- rel_out = rel_repr @ w_rel ----------------
__global__ void k_rel_out(const float* rel_repr, const __bf16* WT_rel, float* out2) {
    __shared__ __align__(16) __bf16 As[16 * AS_STR];
    const int mbase = blockIdx.x * 16;
    const int tid   = threadIdx.x;               // 128
    for (int i = tid; i < 16 * D; i += 128) {
        const int m = i >> 8, k = i & 255;
        As[m * AS_STR + k] = (__bf16)rel_repr[(size_t)(mbase + m) * D + k];
    }
    __syncthreads();
    const int wave = tid >> 5, lane = tid & 31;
    const int n0   = wave * 64;
    v8f acc[4] = {};
#pragma unroll
    for (int k0 = 0; k0 < D; k0 += 32) {
        const v16bf a = load_a_lds(As, AS_STR, k0, lane);
#pragma unroll
        for (int nt = 0; nt < 4; ++nt) {
            const v16bf b = load_b_nt(WT_rel, D, n0 + nt * 16, k0, lane);
            acc[nt] = wmma_bf16(a, b, acc[nt]);
        }
    }
    const int m0 = (lane < 16) ? 0 : 8;
#pragma unroll
    for (int nt = 0; nt < 4; ++nt) {
        const int n = n0 + nt * 16 + (lane & 15);
#pragma unroll
        for (int r = 0; r < 8; ++r)
            out2[(size_t)(mbase + m0 + r) * D + n] = acc[nt][r];
    }
}

// ---------------- BatchNorm (batch stats over 80000 rows) ----------------
__global__ void k_col_stats(const float* out, float* colsum, float* colsumsq) {
    const int c  = threadIdx.x;                  // 256 threads, grid 625
    const int r0 = blockIdx.x * 128;
    float s = 0.f, ss = 0.f;
    for (int i = 0; i < 128; ++i) {
        const float v = out[(size_t)(r0 + i) * D + c];
        s += v; ss += v * v;
    }
    atomicAdd(&colsum[c], s);
    atomicAdd(&colsumsq[c], ss);
}

__global__ void k_bn_apply(float* out, const float* colsum, const float* colsumsq) {
    const int c = threadIdx.x;                   // row stride == blockDim == 256
    const size_t idx = (size_t)blockIdx.x * D + c;
    const float mean = colsum[c] * (1.f / (float)NNODES);
    const float var  = colsumsq[c] * (1.f / (float)NNODES) - mean * mean;
    out[idx] = (out[idx] - mean) * rsqrtf(var + 1e-5f);
}

// ---------------- host launcher ----------------
extern "C" void kernel_launch(void* const* d_in, const int* in_sizes, int n_in,
                              void* d_out, int out_size, void* d_ws, size_t ws_size,
                              hipStream_t stream) {
    const float* x        = (const float*)d_in[0];
    const float* rel_repr = (const float*)d_in[1];
    const int*   esrc     = (const int*)  d_in[2];
    const int*   edst     = (const int*)  d_in[3];
    const int*   etype    = (const int*)  d_in[4];
    const float* enorm    = (const float*)d_in[5];
    const float* in_w     = (const float*)d_in[6];
    const float* out_w    = (const float*)d_in[7];
    const float* loop_w   = (const float*)d_in[8];
    const float* w_rel    = (const float*)d_in[9];
    const float* loop_rel = (const float*)d_in[10];
    const float* bias     = (const float*)d_in[11];

    float* out  = (float*)d_out;                       // [80000,256]
    float* out2 = (float*)d_out + (size_t)NNODES * D;  // [400,256]

    // workspace carve-up
    char* w = (char*)d_ws;
    __bf16* PT = (__bf16*)w;                 w += (size_t)(NBINS + 1) * MAT_EL * 2; // 801 composites
    __bf16* WT = (__bf16*)w;                 w += (size_t)4 * MAT_EL * 2;           // 4 transposed W
    __bf16* Xb = (__bf16*)w;                 w += (size_t)NNODES * D * 2;           // bf16 copy of x
    int* sorted = (int*)w;                   w += (size_t)NEDGES * 4;
    int* meta   = (int*)w;                   // hist|offsets|cursor|tileoff|colsum|colsumsq
    int*   hist     = meta;
    int*   offsets  = meta + NBINS;
    int*   cursor   = meta + NBINS + (NBINS + 1);
    int*   tileoff  = meta + 2 * NBINS + (NBINS + 1);
    float* colsum   = (float*)(meta + 2 * NBINS + 2 * (NBINS + 1));
    float* colsumsq = colsum + D;
    const long long META32 = 2 * NBINS + 2 * (NBINS + 1) + 2 * D;   // 3714

    // zero: aggregation target + metadata
    {
        const long long n = (long long)NNODES * D;
        k_zero32<<<dim3((unsigned)((n + 255) / 256)), dim3(256), 0, stream>>>((unsigned*)out, n);
        k_zero32<<<dim3((unsigned)((META32 + 255) / 256)), dim3(256), 0, stream>>>((unsigned*)meta, META32);
    }
    // bf16 copy of x (halves per-edge gather traffic; enables raw async DMA gather)
    k_x_to_bf16<<<dim3(NNODES * (D / 4) / 256), dim3(256), 0, stream>>>(x, Xb);
    // precompute transposed weights + 801 composite matrices (WMMA)
    k_transpose_w<<<dim3(4 * 256), dim3(256), 0, stream>>>(in_w, out_w, loop_w, w_rel, WT);
    k_build_composite<<<dim3(NBINS + 1), dim3(256), 0, stream>>>(rel_repr, loop_rel, WT, PT);
    // counting sort by (dir, type)
    k_hist<<<dim3((NEDGES + 255) / 256), dim3(256), 0, stream>>>(etype, hist);
    k_scan<<<dim3(1), dim3(1024), 0, stream>>>(hist, offsets, cursor, tileoff);
    k_scatter_idx<<<dim3((NEDGES + 255) / 256), dim3(256), 0, stream>>>(etype, cursor, sorted);
    // main message GEMM + scatter
    k_edge_gemm<<<dim3(MAX_TILES), dim3(128), 0, stream>>>(Xb, esrc, edst, enorm, sorted,
                                                           offsets, tileoff, PT, out);
    // self-loop + bias
    k_loop_gemm<<<dim3(NNODES / 16), dim3(128), 0, stream>>>(Xb, PT + (size_t)NBINS * MAT_EL,
                                                             bias, out);
    // relation output
    k_rel_out<<<dim3(NREL / 16), dim3(128), 0, stream>>>(rel_repr, WT + 3 * MAT_EL, out2);
    // batchnorm
    k_col_stats<<<dim3(NNODES / 128), dim3(256), 0, stream>>>(out, colsum, colsumsq);
    k_bn_apply<<<dim3(NNODES), dim3(256), 0, stream>>>(out, colsum, colsumsq);
}
